// BiaffineChart_8065948581892
// MI455X (gfx1250) — matrix operand order repaired and checked by reading, not compiled, not run on hardware
//
#include <hip/hip_runtime.h>
#include <hip/hip_bf16.h>

#define N_TOK 4096
#define D_DIM 1024

typedef __bf16 bf16_t;
typedef bf16_t v16bf __attribute__((ext_vector_type(16)));
typedef bf16_t v8bf  __attribute__((ext_vector_type(8)));
typedef float  v8f   __attribute__((ext_vector_type(8)));

// ---------------- elementwise f32 -> bf16 ----------------
__global__ void cvt_bf16_kernel(const float* __restrict__ in, bf16_t* __restrict__ out, int n) {
  int i = blockIdx.x * blockDim.x + threadIdx.x;
  if (i < n) out[i] = (bf16_t)in[i];
}

// ------- transpose+convert matrix [l][r] f32 -> [r][l] bf16 (B wants K-contiguous) -------
__global__ void cvt_transpose_kernel(const float* __restrict__ in, bf16_t* __restrict__ out) {
  __shared__ bf16_t tile[32][33];
  const int bx = blockIdx.x * 32;          // r range (read cols)
  const int by = blockIdx.y * 32;          // l range (read rows)
  const int tx = threadIdx.x, ty = threadIdx.y;   // 32 x 8
#pragma unroll
  for (int j = 0; j < 32; j += 8)
    tile[ty + j][tx] = (bf16_t)in[(size_t)(by + ty + j) * D_DIM + (bx + tx)];
  __syncthreads();
#pragma unroll
  for (int j = 0; j < 32; j += 8)
    out[(size_t)(bx + ty + j) * D_DIM + (by + tx)] = tile[tx][ty + j];
}

// ------- lin[i] = dot(x[i,:], w) + b  (one wave32 per row, f32) -------
__global__ void lin_kernel(const float* __restrict__ x, const float* __restrict__ w,
                           const float* __restrict__ b, float* __restrict__ out) {
  const int row  = blockIdx.x * (blockDim.x >> 5) + (threadIdx.x >> 5);
  const int lane = threadIdx.x & 31;
  const float* xr = x + (size_t)row * D_DIM;
  float s = 0.f;
#pragma unroll 4
  for (int k = lane; k < D_DIM; k += 32) s += xr[k] * w[k];
#pragma unroll
  for (int off = 16; off > 0; off >>= 1) s += __shfl_xor(s, off, 32);
  if (lane == 0) out[row] = s + b[0];
}

// ---------------- WMMA fragment loaders (layouts per CDNA5 ISA) ----------------
// A 16x32 bf16, row-major [M,K]: lane L -> row (L&15); K = kb..kb+7 (v0-3), kb+16..kb+23 (v4-7), kb=(L>>4)*8
__device__ __forceinline__ v16bf load_a_frag(const bf16_t* __restrict__ tile, int ld, int lane) {
  const bf16_t* p = tile + (size_t)(lane & 15) * ld + ((lane >> 4) << 3);
  union { v16bf v; v8bf h[2]; } u;
  u.h[0] = *(const v8bf*)(p);
  u.h[1] = *(const v8bf*)(p + 16);
  return u.v;
}
// B 32x16 bf16 from K-contiguous storage [N,K]: lane L -> col (L&15); K = kb..kb+7 (v0-3), kb+16..kb+23 (v4-7), kb=(L>>4)*8
__device__ __forceinline__ v16bf load_b_frag(const bf16_t* __restrict__ tile, int ld, int lane) {
  const bf16_t* p = tile + (size_t)(lane & 15) * ld + ((lane >> 4) << 3);
  union { v16bf v; v8bf h[2]; } u;
  u.h[0] = *(const v8bf*)(p);
  u.h[1] = *(const v8bf*)(p + 16);
  return u.v;
}

// ---------------- GEMM 1: xm = xl @ M   (bf16 in, f32 acc, bf16 out) ----------------
__global__ __launch_bounds__(128)
void gemm_xm_kernel(const bf16_t* __restrict__ A,   // xl_bf [N_TOK, D]
                    const bf16_t* __restrict__ Bt,  // mT_bf [D(r), D(l)]  (K-contiguous)
                    bf16_t* __restrict__ Xm) {      // [N_TOK, D]
  const int lane = threadIdx.x & 31;
  const int wave = threadIdx.x >> 5;
  const int row0 = blockIdx.y * 128 + (wave >> 1) * 64;
  const int col0 = blockIdx.x * 128 + (wave & 1) * 64;

  v8f acc[4][4];
  v8f z = {};
#pragma unroll
  for (int i = 0; i < 4; ++i)
#pragma unroll
    for (int j = 0; j < 4; ++j) acc[i][j] = z;

  for (int k0 = 0; k0 < D_DIM; k0 += 32) {
    v16bf a[4], b[4];
#pragma unroll
    for (int t = 0; t < 4; ++t)
      a[t] = load_a_frag(A + (size_t)(row0 + 16 * t) * D_DIM + k0, D_DIM, lane);
#pragma unroll
    for (int t = 0; t < 4; ++t)
      b[t] = load_b_frag(Bt + (size_t)(col0 + 16 * t) * D_DIM + k0, D_DIM, lane);
#pragma unroll
    for (int i = 0; i < 4; ++i)
#pragma unroll
      for (int j = 0; j < 4; ++j)
        acc[i][j] = __builtin_amdgcn_wmma_f32_16x16x32_bf16(
            false, a[i], false, b[j], (short)0, acc[i][j], false, false);
  }

  // C layout: vgpr r -> M = r + 8*(lane>>4); N = lane&15
  const int mhi = (lane >> 4) << 3;
  const int nlo = lane & 15;
#pragma unroll
  for (int i = 0; i < 4; ++i)
#pragma unroll
    for (int j = 0; j < 4; ++j) {
      const int n = col0 + j * 16 + nlo;
#pragma unroll
      for (int r = 0; r < 8; ++r) {
        const int m = row0 + i * 16 + mhi + r;
        Xm[(size_t)m * D_DIM + n] = (bf16_t)acc[i][j][r];
      }
    }
}

// ------- GEMM 2 (fused): x = relu(xm @ xr^T + bias + linl[i] + linr[j]); scores = mask?x:0 -------
__global__ __launch_bounds__(128)
void biaffine_main_kernel(const bf16_t* __restrict__ A,   // xm_bf [N_TOK, D]
                          const bf16_t* __restrict__ B,   // xr_bf [N_TOK, D] (K-contiguous for B frags)
                          const float* __restrict__ linl,
                          const float* __restrict__ linr,
                          const float* __restrict__ bias,
                          const unsigned char* __restrict__ mask,
                          float* __restrict__ scores,
                          float* __restrict__ xout) {
  const int lane = threadIdx.x & 31;
  const int wave = threadIdx.x >> 5;
  const int row0 = blockIdx.y * 128 + (wave >> 1) * 64;
  const int col0 = blockIdx.x * 128 + (wave & 1) * 64;

  v8f acc[4][4];
  v8f z = {};
#pragma unroll
  for (int i = 0; i < 4; ++i)
#pragma unroll
    for (int j = 0; j < 4; ++j) acc[i][j] = z;

  for (int k0 = 0; k0 < D_DIM; k0 += 32) {
    v16bf a[4], b[4];
#pragma unroll
    for (int t = 0; t < 4; ++t)
      a[t] = load_a_frag(A + (size_t)(row0 + 16 * t) * D_DIM + k0, D_DIM, lane);
#pragma unroll
    for (int t = 0; t < 4; ++t)
      b[t] = load_b_frag(B + (size_t)(col0 + 16 * t) * D_DIM + k0, D_DIM, lane);
#pragma unroll
    for (int i = 0; i < 4; ++i)
#pragma unroll
      for (int j = 0; j < 4; ++j)
        acc[i][j] = __builtin_amdgcn_wmma_f32_16x16x32_bf16(
            false, a[i], false, b[j], (short)0, acc[i][j], false, false);
  }

  const float bv = bias[0];
  const int mhi = (lane >> 4) << 3;
  const int nlo = lane & 15;
#pragma unroll
  for (int i = 0; i < 4; ++i) {
    float ll[8];
#pragma unroll
    for (int r = 0; r < 8; ++r) ll[r] = linl[row0 + i * 16 + mhi + r];
#pragma unroll
    for (int j = 0; j < 4; ++j) {
      const int n = col0 + j * 16 + nlo;
      const float lr = linr[n] + bv;
#pragma unroll
      for (int r = 0; r < 8; ++r) {
        const int m = row0 + i * 16 + mhi + r;
        float v = acc[i][j][r] + ll[r] + lr;
        v = v > 0.f ? v : 0.f;                       // relu
        const size_t idx = (size_t)m * N_TOK + n;
        xout[idx] = v;
        scores[idx] = mask[idx] ? v : 0.f;           // where(mask, x, 0)
      }
    }
  }
}

extern "C" void kernel_launch(void* const* d_in, const int* in_sizes, int n_in,
                              void* d_out, int out_size, void* d_ws, size_t ws_size,
                              hipStream_t stream) {
  (void)in_sizes; (void)n_in; (void)out_size; (void)ws_size;
  const float*         x_l    = (const float*)d_in[0];
  const float*         x_r    = (const float*)d_in[1];
  const unsigned char* mask   = (const unsigned char*)d_in[2];  // jnp.bool_ layout (1 byte)
  const float*         matrix = (const float*)d_in[3];
  const float*         bias   = (const float*)d_in[4];
  const float*         wl     = (const float*)d_in[5];
  const float*         bl     = (const float*)d_in[6];
  const float*         wr     = (const float*)d_in[7];
  const float*         br     = (const float*)d_in[8];
  // d_in[9] = s_ind, unused

  // workspace carve-up (~26.3 MB)
  char* ws = (char*)d_ws;
  bf16_t* xl_bf = (bf16_t*)ws;
  bf16_t* xr_bf = xl_bf + (size_t)N_TOK * D_DIM;
  bf16_t* mT_bf = xr_bf + (size_t)N_TOK * D_DIM;
  bf16_t* xm_bf = mT_bf + (size_t)D_DIM * D_DIM;
  float*  linl  = (float*)(xm_bf + (size_t)N_TOK * D_DIM);
  float*  linr  = linl + N_TOK;

  float* scores = (float*)d_out;                       // output 0
  float* xout   = scores + (size_t)N_TOK * N_TOK;      // output 1

  const int nElems = N_TOK * D_DIM;

  // 1) precision conversions
  cvt_bf16_kernel<<<(nElems + 255) / 256, 256, 0, stream>>>(x_l, xl_bf, nElems);
  cvt_bf16_kernel<<<(nElems + 255) / 256, 256, 0, stream>>>(x_r, xr_bf, nElems);
  cvt_transpose_kernel<<<dim3(D_DIM / 32, D_DIM / 32), dim3(32, 8), 0, stream>>>(matrix, mT_bf);

  // 2) linear terms (f32)
  lin_kernel<<<N_TOK / 8, 256, 0, stream>>>(x_l, wl, bl, linl);
  lin_kernel<<<N_TOK / 8, 256, 0, stream>>>(x_r, wr, br, linr);

  // 3) xm = xl @ M   (4096x1024x1024, bf16 WMMA)
  gemm_xm_kernel<<<dim3(D_DIM / 128, N_TOK / 128), 128, 0, stream>>>(xl_bf, mT_bf, xm_bf);

  // 4) fused biaffine GEMM + epilogue (4096x4096x1024, bf16 WMMA)
  biaffine_main_kernel<<<dim3(N_TOK / 128, N_TOK / 128), 128, 0, stream>>>(
      xm_bf, xr_bf, linl, linr, bias, mask, scores, xout);
}
